// MistralModel_83803401880087
// MI455X (gfx1250) — matrix operand order repaired and checked by reading, not compiled, not run on hardware
//
#include <hip/hip_runtime.h>

// ---------------- constants (match reference) ----------------
#define S_   2048
#define H_   2048
#define NH_  16
#define NKV_ 8
#define HD_  128
#define I_   5632
#define L_   2
#define WIN_ 1024
#define EPS_ 1e-6f

typedef __attribute__((ext_vector_type(16))) _Float16 v16h;
typedef __attribute__((ext_vector_type(8)))  _Float16 v8h;
typedef __attribute__((ext_vector_type(4)))  _Float16 v4h;
typedef __attribute__((ext_vector_type(8)))  float    v8f;
typedef __attribute__((ext_vector_type(4)))  float    v4f;
typedef __attribute__((ext_vector_type(4)))  unsigned int u32x4;
typedef __attribute__((ext_vector_type(8)))  int      i32x8;
typedef __attribute__((ext_vector_type(4)))  int      i32x4;

union FragA { v16h v; v8h h[2]; };

__device__ __forceinline__ v8f vzero8() {
  v8f z = {0.f,0.f,0.f,0.f,0.f,0.f,0.f,0.f};
  return z;
}

template<int M>
__device__ __forceinline__ float swz_xor(float x) {
  return __int_as_float(__builtin_amdgcn_ds_swizzle(__float_as_int(x), 0x1f | (M << 10)));
}
__device__ __forceinline__ float half_reduce_max(float x) {
  x = fmaxf(x, swz_xor<1>(x));
  x = fmaxf(x, swz_xor<2>(x));
  x = fmaxf(x, swz_xor<4>(x));
  x = fmaxf(x, swz_xor<8>(x));
  return x;
}
__device__ __forceinline__ float half_reduce_sum(float x) {
  x += swz_xor<1>(x);
  x += swz_xor<2>(x);
  x += swz_xor<4>(x);
  x += swz_xor<8>(x);
  return x;
}

// ---------------- Tensor Data Mover: 2D tile (128 rows x 32 f16) -> LDS ----------------
// D# layout per CDNA5 ISA ch.8: group0 = {count/lds_addr/global_addr/type},
// group1 = {data_size, tensor dims, tile dims, strides}. 2D tensor -> groups 2/3 zero.
__device__ __forceinline__ void tdm_load_tile_f16(unsigned int lds_byte_off,
                                                  const _Float16* gsrc,
                                                  int row_stride_elems) {
  unsigned long long ga = (unsigned long long)(uintptr_t)gsrc;
  unsigned int K = (unsigned int)row_stride_elems;
  u32x4 g0;
  g0[0] = 1u;                                               // count=1, user mode
  g0[1] = lds_byte_off;                                     // lds_addr
  g0[2] = (unsigned int)ga;                                 // global_addr[31:0]
  g0[3] = ((unsigned int)(ga >> 32) & 0x01FFFFFFu) | (2u << 30);  // addr[56:32] | type=2
  i32x8 g1;
  g1[0] = (int)(1u << 16);                    // workgroup_mask=0, data_size=1 (2 bytes)
  g1[1] = (int)((K & 0xFFFFu) << 16);         // tensor_dim0[15:0] @ bits 63:48
  g1[2] = (int)((K >> 16) | (128u << 16));    // tensor_dim0[31:16] | tensor_dim1[15:0]=128
  g1[3] = (int)(32u << 16);                   // tensor_dim1[31:16]=0 | tile_dim0=32
  g1[4] = (int)128u;                          // tile_dim1=128 | tile_dim2=0
  g1[5] = (int)K;                             // tensor_dim0_stride[31:0]
  g1[6] = 0;                                  // stride0[47:32]=0 | stride1[15:0]=0
  g1[7] = 0;
  i32x4 z4 = {0, 0, 0, 0};
#if __clang_major__ >= 23
  i32x8 z8 = {0, 0, 0, 0, 0, 0, 0, 0};
  __builtin_amdgcn_tensor_load_to_lds(g0, g1, z4, z4, z8, 0);
#else
  __builtin_amdgcn_tensor_load_to_lds(g0, g1, z4, z4, 0);
#endif
}

// ---------------- embedding gather ----------------
__global__ __launch_bounds__(256) void gather_kernel(const int* __restrict__ ids,
                                                     const float* __restrict__ embed,
                                                     float* __restrict__ x) {
  int s = blockIdx.x;
  int id = ids[s];
  const float* src = embed + (size_t)id * H_;
  float* dst = x + (size_t)s * H_;
  for (int c = threadIdx.x; c < H_; c += 256) dst[c] = src[c];
}

// ---------------- RMSNorm (f32 out, for final) ----------------
__global__ __launch_bounds__(256) void rmsnorm_kernel(const float* __restrict__ x,
                                                      const float* __restrict__ w,
                                                      float* __restrict__ out) {
  int row = blockIdx.x;
  const float* xr = x + (size_t)row * H_;
  float ss = 0.f;
  for (int c = threadIdx.x; c < H_; c += 256) { float t = xr[c]; ss += t * t; }
  __shared__ float red[256];
  red[threadIdx.x] = ss;
  __syncthreads();
  for (int st = 128; st > 0; st >>= 1) {
    if (threadIdx.x < st) red[threadIdx.x] += red[threadIdx.x + st];
    __syncthreads();
  }
  float rs = rsqrtf(red[0] * (1.0f / (float)H_) + EPS_);
  float* o = out + (size_t)row * H_;
  for (int c = threadIdx.x; c < H_; c += 256) o[c] = xr[c] * rs * w[c];
}

// ---------------- RMSNorm (f16 out, feeds GEMM A operand) ----------------
__global__ __launch_bounds__(256) void rmsnorm_f16_kernel(const float* __restrict__ x,
                                                          const float* __restrict__ w,
                                                          _Float16* __restrict__ out) {
  int row = blockIdx.x;
  const float* xr = x + (size_t)row * H_;
  float ss = 0.f;
  for (int c = threadIdx.x; c < H_; c += 256) { float t = xr[c]; ss += t * t; }
  __shared__ float red[256];
  red[threadIdx.x] = ss;
  __syncthreads();
  for (int st = 128; st > 0; st >>= 1) {
    if (threadIdx.x < st) red[threadIdx.x] += red[threadIdx.x + st];
    __syncthreads();
  }
  float rs = rsqrtf(red[0] * (1.0f / (float)H_) + EPS_);
  _Float16* o = out + (size_t)row * H_;
  for (int c = threadIdx.x; c < H_; c += 256) o[c] = (_Float16)(xr[c] * rs * w[c]);
}

// ---------------- f32 [K][N] -> f16 [N][K] transpose ----------------
__global__ __launch_bounds__(256) void transpose_kernel(const float* __restrict__ in,
                                                        _Float16* __restrict__ out,
                                                        int K, int N) {
  __shared__ float tile[32][33];
  int bx = blockIdx.x * 32;  // N
  int by = blockIdx.y * 32;  // K
  int tx = threadIdx.x, ty = threadIdx.y;  // 32 x 8
  for (int r = 0; r < 32; r += 8)
    tile[ty + r][tx] = in[(size_t)(by + ty + r) * N + bx + tx];
  __syncthreads();
  for (int r = 0; r < 32; r += 8)
    out[(size_t)(bx + ty + r) * K + by + tx] = (_Float16)tile[tx][ty + r];
}

// ---------------- v [S][NKV*HD] f32 -> vt [NKV][HD][S] f16 ----------------
__global__ __launch_bounds__(256) void vtranspose_kernel(const float* __restrict__ v,
                                                         _Float16* __restrict__ vt) {
  __shared__ float tile[32][33];
  int s0 = blockIdx.x * 32, d0 = blockIdx.y * 32, h = blockIdx.z;
  int tx = threadIdx.x, ty = threadIdx.y;
  for (int r = 0; r < 32; r += 8)
    tile[ty + r][tx] = v[(size_t)(s0 + ty + r) * (NKV_ * HD_) + h * HD_ + d0 + tx];
  __syncthreads();
  for (int r = 0; r < 32; r += 8)
    vt[((size_t)h * HD_ + d0 + ty + r) * S_ + s0 + tx] = (_Float16)tile[tx][ty + r];
}

// ---------------- RoPE in-place (q or k) ----------------
__global__ __launch_bounds__(256) void rope_kernel(float* __restrict__ q, int nheads) {
  int idx = blockIdx.x * 256 + threadIdx.x;
  int total = S_ * nheads * (HD_ / 2);
  if (idx >= total) return;
  int j = idx & 63;
  int t = idx >> 6;
  int hh = t % nheads;
  int s = t / nheads;
  float inv = __powf(10000.0f, -(float)j / 64.0f);
  float ang = (float)s * inv;
  float sn, cs;
  __sincosf(ang, &sn, &cs);
  float* base = q + ((size_t)s * nheads + hh) * HD_;
  float a = base[j], b = base[j + 64];
  base[j]      = a * cs - b * sn;
  base[j + 64] = b * cs + a * sn;
}

// ---------------- SiLU(g) * u -> f16 out ----------------
__global__ __launch_bounds__(256) void silu_mul_kernel(const float* __restrict__ g,
                                                       const float* __restrict__ u,
                                                       _Float16* __restrict__ out,
                                                       int n) {
  int i = blockIdx.x * 256 + threadIdx.x;
  if (i < n) {
    float gv = g[i];
    float sv = gv / (1.0f + __expf(-gv));
    out[i] = (_Float16)(sv * u[i]);
  }
}

// ---------------- WMMA GEMM with TDM double-buffered staging ----------------
// C[M][N](f32) = A16[M][K](f16) x Bt[N][K](f16)  (+ Cin residual)
#define BM 128
#define BN 128
#define BK 32

__global__ __launch_bounds__(256) void gemm_kernel(const _Float16* __restrict__ A16,
                                                   const _Float16* __restrict__ Bt,
                                                   const float* __restrict__ Cin,
                                                   float* __restrict__ Cout,
                                                   int M, int N, int K) {
  __shared__ _Float16 As[2][BM][BK];
  __shared__ _Float16 Bs[2][BM][BK];
  const int tid = threadIdx.x;
  const int lane = tid & 31;
  const int wave = tid >> 5;   // 0..7
  const int wm = wave >> 1;    // 0..3  (M offset wm*32)
  const int wn = wave & 1;     // 0..1  (N offset wn*64)
  const int bm0 = blockIdx.y * BM;
  const int bn0 = blockIdx.x * BN;
  const int ln = lane & 15;
  const int kb = (lane & 16) ? 16 : 0;    // A-fragment K split
  const int kboff = (lane & 16) ? 16 : 0; // B-fragment K base

  v8f acc[2][4];
  for (int i = 0; i < 2; i++)
    for (int j = 0; j < 4; j++) acc[i][j] = vzero8();

  const unsigned int aOff[2] = { (unsigned int)(uintptr_t)&As[0][0][0],
                                 (unsigned int)(uintptr_t)&As[1][0][0] };
  const unsigned int bOff[2] = { (unsigned int)(uintptr_t)&Bs[0][0][0],
                                 (unsigned int)(uintptr_t)&Bs[1][0][0] };
  const _Float16* aBase = A16 + (size_t)bm0 * K;
  const _Float16* bBase = Bt + (size_t)bn0 * K;
  const int nk = K / BK;

  if (wave == 0) {
    tdm_load_tile_f16(aOff[0], aBase, K);
    tdm_load_tile_f16(bOff[0], bBase, K);
  }

  for (int i = 0; i < nk; ++i) {
    const int cur = i & 1;
    if (wave == 0) {
      if (i + 1 < nk) {
        tdm_load_tile_f16(aOff[1 - cur], aBase + (i + 1) * BK, K);
        tdm_load_tile_f16(bOff[1 - cur], bBase + (i + 1) * BK, K);
        __builtin_amdgcn_s_wait_tensorcnt(2);   // stage i complete, stage i+1 in flight
      } else {
        __builtin_amdgcn_s_wait_tensorcnt(0);
      }
    }
    __syncthreads();

    FragA a[2];
    v16h b[4];
    #pragma unroll
    for (int ii = 0; ii < 2; ii++) {
      int row = wm * 32 + ii * 16 + ln;
      a[ii].h[0] = *(const v8h*)&As[cur][row][kb];
      a[ii].h[1] = *(const v8h*)&As[cur][row][8 + kb];
    }
    #pragma unroll
    for (int j = 0; j < 4; j++) {
      int row = wn * 64 + j * 16 + ln;
      b[j] = *(const v16h*)&Bs[cur][row][kboff];
    }
    #pragma unroll
    for (int ii = 0; ii < 2; ii++)
      #pragma unroll
      for (int j = 0; j < 4; j++)
        acc[ii][j] = __builtin_amdgcn_wmma_f32_16x16x32_f16(
            false, a[ii].v, false, b[j], (short)0, acc[ii][j], false, false);
    __syncthreads();
  }

  const int mo = (lane & 16) ? 8 : 0;
  for (int i = 0; i < 2; i++) {
    for (int j = 0; j < 4; j++) {
      #pragma unroll
      for (int vv = 0; vv < 8; vv++) {
        int gm = bm0 + wm * 32 + i * 16 + mo + vv;
        int gn = bn0 + wn * 64 + j * 16 + ln;
        size_t off = (size_t)gm * N + gn;
        float val = acc[i][j][vv];
        if (Cin) val += Cin[off];
        Cout[off] = val;
      }
    }
  }
}

// ---------------- flash attention (sliding-window causal, GQA) ----------------
#define NEG_BIG  (-1e30f)
#define NEG_GRD  (-0.5e30f)

__global__ __launch_bounds__(128) void attn_kernel(const float* __restrict__ q,
                                                   const float* __restrict__ k,
                                                   const _Float16* __restrict__ vt,
                                                   _Float16* __restrict__ o) {
  const int lane = threadIdx.x & 31;
  const int wave = threadIdx.x >> 5;          // 0..3
  const int q0 = blockIdx.x * 16;
  const int head = blockIdx.y * 4 + wave;
  const int kvh = head >> 1;                  // NH/NKV = 2
  const int ln = lane & 15;
  const int kb = (lane & 16) ? 16 : 0;
  const int kboff = (lane & 16) ? 16 : 0;
  const int mo = (lane & 16) ? 8 : 0;
  const float scale = 0.08838834764831845f;   // 1/sqrt(128)

  __shared__ _Float16 pbuf[4][16][32];

  // Q fragments (A-layout), pre-scaled
  FragA qa[4];
  {
    const float* qrow = q + ((size_t)(q0 + ln) * NH_ + head) * HD_;
    #pragma unroll
    for (int kt = 0; kt < 4; kt++) {
      #pragma unroll
      for (int i = 0; i < 8; i++)
        qa[kt].h[0][i] = (_Float16)(qrow[kt * 32 + kb + i] * scale);
      #pragma unroll
      for (int i = 0; i < 8; i++)
        qa[kt].h[1][i] = (_Float16)(qrow[kt * 32 + 8 + kb + i] * scale);
    }
  }

  v8f acc[8];
  for (int t = 0; t < 8; t++) acc[t] = vzero8();
  float mstat[8], lstat[8];
  for (int vv = 0; vv < 8; vv++) { mstat[vv] = NEG_BIG; lstat[vv] = 0.f; }

  const int kstart = (q0 > WIN_) ? ((q0 - WIN_) & ~31) : 0;
  const int kend = q0 + 15;

  for (int k0 = kstart; k0 <= kend; k0 += 32) {
    v8f s[2];
    #pragma unroll
    for (int jsub = 0; jsub < 2; jsub++) {
      int kt0 = k0 + jsub * 16;
      v8f sc = vzero8();
      const float* krow = k + ((size_t)(kt0 + ln) * NKV_ + kvh) * HD_;
      #pragma unroll
      for (int kt = 0; kt < 4; kt++) {
        v16h kf;
        #pragma unroll
        for (int i = 0; i < 16; i++)
          kf[i] = (_Float16)krow[kt * 32 + kboff + i];
        sc = __builtin_amdgcn_wmma_f32_16x16x32_f16(
            false, qa[kt].v, false, kf, (short)0, sc, false, false);
      }
      int kp = kt0 + ln;
      #pragma unroll
      for (int vv = 0; vv < 8; vv++) {
        int qp = q0 + mo + vv;
        bool ok = (kp <= qp) && (kp >= qp - WIN_);
        sc[vv] = ok ? sc[vv] : NEG_BIG;
      }
      s[jsub] = sc;
    }

    // online softmax update
    v8f p0, p1;
    #pragma unroll
    for (int vv = 0; vv < 8; vv++) {
      float rm = half_reduce_max(fmaxf(s[0][vv], s[1][vv]));
      float mn = fmaxf(mstat[vv], rm);
      float alpha = (mn <= NEG_GRD) ? 1.0f : __expf(mstat[vv] - mn);
      float e0 = (s[0][vv] <= NEG_GRD) ? 0.0f : __expf(s[0][vv] - mn);
      float e1 = (s[1][vv] <= NEG_GRD) ? 0.0f : __expf(s[1][vv] - mn);
      p0[vv] = e0;
      p1[vv] = e1;
      float rs = half_reduce_sum(e0 + e1);
      lstat[vv] = lstat[vv] * alpha + rs;
      mstat[vv] = mn;
      #pragma unroll
      for (int t = 0; t < 8; t++) acc[t][vv] *= alpha;
    }

    // C-layout P -> A-layout fragment via per-wave LDS bounce
    #pragma unroll
    for (int vv = 0; vv < 8; vv++) {
      pbuf[wave][mo + vv][ln]      = (_Float16)p0[vv];
      pbuf[wave][mo + vv][16 + ln] = (_Float16)p1[vv];
    }
    asm volatile("" ::: "memory");
    FragA pa;
    pa.h[0] = *(const v8h*)&pbuf[wave][ln][kb];
    pa.h[1] = *(const v8h*)&pbuf[wave][ln][8 + kb];
    asm volatile("" ::: "memory");

    // O += P(16x32) x V(32x128): 8 WMMAs, B from f16 V^T (contiguous in key dim)
    #pragma unroll
    for (int t = 0; t < 8; t++) {
      const _Float16* vr = vt + ((size_t)kvh * HD_ + t * 16 + ln) * S_ + k0;
      v16h vb = *(const v16h*)(vr + kboff);
      acc[t] = __builtin_amdgcn_wmma_f32_16x16x32_f16(
          false, pa.v, false, vb, (short)0, acc[t], false, false);
    }
  }

  // epilogue: divide by softmax sum, write f16 [S][NH*HD]
  #pragma unroll
  for (int t = 0; t < 8; t++) {
    #pragma unroll
    for (int vv = 0; vv < 8; vv++) {
      float val = acc[t][vv] / lstat[vv];
      o[((size_t)(q0 + mo + vv) * NH_ + head) * HD_ + t * 16 + ln] = (_Float16)val;
    }
  }
}

// ---------------- launch ----------------
extern "C" void kernel_launch(void* const* d_in, const int* in_sizes, int n_in,
                              void* d_out, int out_size, void* d_ws, size_t ws_size,
                              hipStream_t stream) {
  (void)in_sizes; (void)n_in; (void)out_size; (void)ws_size;
  const int*   ids   = (const int*)d_in[0];
  const float* embed = (const float*)d_in[1];
  const float* Wq = (const float*)d_in[2];
  const float* Wk = (const float*)d_in[3];
  const float* Wv = (const float*)d_in[4];
  const float* Wo = (const float*)d_in[5];
  const float* Wg = (const float*)d_in[6];
  const float* Wu = (const float*)d_in[7];
  const float* Wd = (const float*)d_in[8];
  const float* ln1 = (const float*)d_in[9];
  const float* ln2 = (const float*)d_in[10];
  const float* normw = (const float*)d_in[11];

  float* x   = (float*)d_ws;
  float* qb  = x + (size_t)S_ * H_;
  float* kbf = qb + (size_t)S_ * NH_ * HD_;
  float* vb  = kbf + (size_t)S_ * NKV_ * HD_;
  float* gb  = vb + (size_t)S_ * NKV_ * HD_;
  float* ub  = gb + (size_t)S_ * I_;
  _Float16* h16  = (_Float16*)(ub + (size_t)S_ * I_);
  _Float16* o16  = h16 + (size_t)S_ * H_;
  _Float16* g16  = o16 + (size_t)S_ * H_;
  _Float16* vt   = g16 + (size_t)S_ * I_;
  _Float16* bt   = vt + (size_t)NKV_ * HD_ * S_;

  dim3 tb(32, 8);

  gather_kernel<<<S_, 256, 0, stream>>>(ids, embed, x);

  for (int l = 0; l < L_; l++) {
    rmsnorm_f16_kernel<<<S_, 256, 0, stream>>>(x, ln1 + (size_t)l * H_, h16);

    // Q = h @ Wq
    transpose_kernel<<<dim3((NH_ * HD_) / 32, H_ / 32), tb, 0, stream>>>(
        Wq + (size_t)l * H_ * NH_ * HD_, bt, H_, NH_ * HD_);
    gemm_kernel<<<dim3((NH_ * HD_) / 128, S_ / 128), 256, 0, stream>>>(
        h16, bt, nullptr, qb, S_, NH_ * HD_, H_);
    // K = h @ Wk
    transpose_kernel<<<dim3((NKV_ * HD_) / 32, H_ / 32), tb, 0, stream>>>(
        Wk + (size_t)l * H_ * NKV_ * HD_, bt, H_, NKV_ * HD_);
    gemm_kernel<<<dim3((NKV_ * HD_) / 128, S_ / 128), 256, 0, stream>>>(
        h16, bt, nullptr, kbf, S_, NKV_ * HD_, H_);
    // V = h @ Wv
    transpose_kernel<<<dim3((NKV_ * HD_) / 32, H_ / 32), tb, 0, stream>>>(
        Wv + (size_t)l * H_ * NKV_ * HD_, bt, H_, NKV_ * HD_);
    gemm_kernel<<<dim3((NKV_ * HD_) / 128, S_ / 128), 256, 0, stream>>>(
        h16, bt, nullptr, vb, S_, NKV_ * HD_, H_);

    rope_kernel<<<(S_ * NH_ * (HD_ / 2)) / 256, 256, 0, stream>>>(qb, NH_);
    rope_kernel<<<(S_ * NKV_ * (HD_ / 2)) / 256, 256, 0, stream>>>(kbf, NKV_);

    vtranspose_kernel<<<dim3(S_ / 32, HD_ / 32, NKV_), tb, 0, stream>>>(vb, vt);

    attn_kernel<<<dim3(S_ / 16, NH_ / 4), 128, 0, stream>>>(qb, kbf, vt, o16);

    // x = x + o @ Wo
    transpose_kernel<<<dim3(H_ / 32, (NH_ * HD_) / 32), tb, 0, stream>>>(
        Wo + (size_t)l * NH_ * HD_ * H_, bt, NH_ * HD_, H_);
    gemm_kernel<<<dim3(H_ / 128, S_ / 128), 256, 0, stream>>>(
        o16, bt, x, x, S_, H_, NH_ * HD_);

    rmsnorm_f16_kernel<<<S_, 256, 0, stream>>>(x, ln2 + (size_t)l * H_, h16);

    // g = h @ Wg ; u = h @ Wu
    transpose_kernel<<<dim3(I_ / 32, H_ / 32), tb, 0, stream>>>(
        Wg + (size_t)l * H_ * I_, bt, H_, I_);
    gemm_kernel<<<dim3(I_ / 128, S_ / 128), 256, 0, stream>>>(
        h16, bt, nullptr, gb, S_, I_, H_);
    transpose_kernel<<<dim3(I_ / 32, H_ / 32), tb, 0, stream>>>(
        Wu + (size_t)l * H_ * I_, bt, H_, I_);
    gemm_kernel<<<dim3(I_ / 128, S_ / 128), 256, 0, stream>>>(
        h16, bt, nullptr, ub, S_, I_, H_);

    silu_mul_kernel<<<(S_ * I_ + 255) / 256, 256, 0, stream>>>(gb, ub, g16, S_ * I_);

    // x = x + gated @ Wd
    transpose_kernel<<<dim3(H_ / 32, I_ / 32), tb, 0, stream>>>(
        Wd + (size_t)l * I_ * H_, bt, I_, H_);
    gemm_kernel<<<dim3(H_ / 128, S_ / 128), 256, 0, stream>>>(
        g16, bt, x, x, S_, H_, I_);
  }

  rmsnorm_kernel<<<S_, 256, 0, stream>>>(x, normw, (float*)d_out);
}
